// KPFP_7842610283226
// MI455X (gfx1250) — matrix operand order repaired and checked by reading, not compiled, not run on hardware
//
#include <hip/hip_runtime.h>

// ---------------------------------------------------------------------------
// KPConv feature propagation for MI455X (gfx1250, wave32, WMMA).
//
//   1) prepw:  W (15x128x128 f32) -> bf16, pre-swizzled into WMMA B-fragment
//              order (one contiguous 32B load per lane in the GEMM).
//   2) knn:    brute-force top-16 (xyz1 batch in LDS, register insertion sort).
//   3) kpconv: per 16-row tile, everything on the matrix pipes:
//        1a: kernel-point weights written straight into bf16 WMMA A-fragments
//            (15x16 per point, S-dim zero-padded to 32).
//        1b: neighbor features gathered to LDS bf16, transposed [n][c][s] so
//            B-fragments are contiguous; padded half reads a zero block.
//        1c: per-point aggregation wf = weights @ feat as 16 WMMA/wave,
//            D scattered into stage-2 A-fragment layout.
//        2 : 16x1920 @ 1920x128 GEMM, 60 WMMA/wave, ReLU, store.
//   4) skip:   copy features2 into out[..., 128:192].
// ---------------------------------------------------------------------------

typedef __attribute__((ext_vector_type(16))) __bf16 v16bf;
typedef __attribute__((ext_vector_type(8)))  float  v8f;

#define BN   4
#define N1   2048
#define N2   8192
#define C1   128
#define C2   64
#define KP   15
#define FO   128
#define SN   16
#define EXT  0.2f
#define KCOLS (KP * C1)      // 1920 flattened (k,c) contraction depth
#define NQ    (KCOLS / 32)   // 60 WMMA K-chunks
#define OUTC  (FO + C2)      // 192
#define ROWS  (BN * N2)      // 32768

// --------------------------- kernel 1: prep W ------------------------------
__global__ __launch_bounds__(256) void prepw_kernel(const float* __restrict__ W,
                                                    __bf16* __restrict__ Bsw) {
  int t = blockIdx.x * 256 + threadIdx.x;          // [0, 60*8*32*16)
  if (t >= NQ * 8 * 32 * 16) return;
  int e    = t & 15;
  int lane = (t >> 4) & 31;
  int fc   = (t >> 9) & 7;
  int q    = t >> 12;
  int khalf = lane >> 4, ncol = lane & 15;
  int kcol  = q * 32 + (e >> 3) * 16 + khalf * 8 + (e & 7);  // flattened (k,c) index
  int k = kcol >> 7, c = kcol & 127;
  int f = fc * 16 + ncol;
  Bsw[t] = (__bf16)W[((size_t)k * C1 + c) * FO + f];
}

// --------------------------- kernel 2: kNN ---------------------------------
__global__ __launch_bounds__(256) void knn_kernel(const float* __restrict__ xyz1,
                                                  const float* __restrict__ xyz2,
                                                  int* __restrict__ idx_out) {
  __shared__ float sx[N1], sy[N1], sz[N1];
  int b = blockIdx.y;
  const float* p1 = xyz1 + (size_t)b * N1 * 3;
  for (int i = threadIdx.x; i < N1; i += 256) {
    sx[i] = p1[i * 3 + 0];
    sy[i] = p1[i * 3 + 1];
    sz[i] = p1[i * 3 + 2];
  }
  __syncthreads();

  int q = blockIdx.x * 256 + threadIdx.x;          // query in [0, N2)
  const float* p2 = xyz2 + ((size_t)b * N2 + q) * 3;
  float qx = p2[0], qy = p2[1], qz = p2[2];

  float bd[SN];
  int   bi[SN];
#pragma unroll
  for (int s = 0; s < SN; ++s) { bd[s] = 3.0e38f; bi[s] = 0; }

  for (int j = 0; j < N1; ++j) {
    float dx = sx[j] - qx, dy = sy[j] - qy, dz = sz[j] - qz;
    float d2 = dx * dx + dy * dy + dz * dz;
    if (d2 < bd[SN - 1]) {                         // rare path: bubble insert
      float cd = d2; int ci = j;
#pragma unroll
      for (int s = 0; s < SN; ++s) {
        if (cd < bd[s]) {
          float td = bd[s]; int ti = bi[s];
          bd[s] = cd; bi[s] = ci;
          cd = td; ci = ti;
        }
      }
    }
  }

  size_t base = ((size_t)b * N2 + q) * SN;
#pragma unroll
  for (int s = 0; s < SN; ++s) idx_out[base + s] = bi[s];
}

// ------------------- kernel 3: fused gather + conv (all-WMMA) --------------
__global__ __launch_bounds__(256) void kpconv_kernel(const float* __restrict__ xyz1,
                                                     const float* __restrict__ feat1,
                                                     const float* __restrict__ xyz2,
                                                     const float* __restrict__ kpts,
                                                     const int*   __restrict__ idx_ws,
                                                     const __bf16* __restrict__ Bsw,
                                                     float* __restrict__ out) {
  // Stage-2 A fragments: [q:60][lane:32][e:16] bf16, 32B contiguous per lane.
  __shared__ __align__(32) __bf16 sA[NQ * 32 * 16];          // 60 KB
  // Neighbor features, transposed bf16: [n:16][c:128][s:16].
  __shared__ __align__(32) __bf16 sF[16 * 128 * 16];         // 64 KB
  // Per-point weight A-fragments (15x16 padded to 16x32): [n:16][lane:32][e:16]
  __shared__ __align__(32) __bf16 swA[16 * 32 * 16];         // 16 KB
  __shared__ __align__(32) __bf16 zpad[16];                  // 32 B of zeros
  __shared__ int   sidx[16 * 16];                            // [n][s]
  __shared__ float skp[16 * 3];                              // kernel points

  int t    = threadIdx.x;
  int row0 = blockIdx.x * 16;          // 16 rows share one batch (N2 % 16 == 0)
  int b    = row0 / N2;
  int n0   = row0 % N2;

  if (t < 48) skp[t] = (t < KP * 3) ? kpts[t] : 0.0f;
  if (t < 16) zpad[t] = (__bf16)0.0f;
  sidx[t] = idx_ws[(size_t)row0 * SN + t];
  __syncthreads();

  // ---- 1a: kernel-point weights -> bf16 A-fragments (thread t -> (n,s)) ----
  {
    int n = t >> 4, s = t & 15;
    int j = sidx[t];
    const float* pq = xyz2 + ((size_t)b * N2 + n0 + n) * 3;
    const float* pn = xyz1 + ((size_t)b * N1 + j) * 3;
    float rx = pn[0] - pq[0], ry = pn[1] - pq[1], rz = pn[2] - pq[2];
    int half = 0;                      // lane offset for this s
    int e    = s;                      // element (A-map: k_col = e)
#pragma unroll
    for (int k = 0; k < 16; ++k) {
      float dx = rx - skp[k * 3 + 0];
      float dy = ry - skp[k * 3 + 1];
      float dz = rz - skp[k * 3 + 2];
      float d  = sqrtf(fmaxf(dx * dx + dy * dy + dz * dz, 1e-12f));
      float w  = fmaxf(1.0f - d / EXT, 0.0f);
      int base = (n * 32 + half + k) * 16;
      swA[base + e]     = (__bf16)((k < KP) ? w : 0.0f);     // s = 0..15
      swA[base + 16 * 16 + e] = (__bf16)0.0f;                // s = 16..31 pad
    }
  }

  // ---- 1b: gather neighbor features -> sF[n][c][s] (bf16, transposed) ----
  {
    int c = t & 127, g = t >> 7;       // g selects s-halves
#pragma unroll
    for (int n = 0; n < 16; ++n) {
#pragma unroll
      for (int ss = 0; ss < 8; ++ss) {
        int s = g * 8 + ss;
        int j = sidx[n * 16 + s];
        float fv = feat1[((size_t)b * N1 + j) * C1 + c];     // coalesced in c
        sF[(n * 128 + c) * 16 + s] = (__bf16)fv;
      }
    }
  }
  __syncthreads();

  // ---- 1c: per-point aggregation wf = weights @ feat via WMMA ----
  // wave w -> points {2w, 2w+1}; D rows = kernel points, cols = channels.
  {
    int wave = t >> 5, lane = t & 31;
    int ncol = lane & 15, mh = lane >> 4;
#pragma unroll
    for (int pp = 0; pp < 2; ++pp) {
      int n = wave * 2 + pp;
      v16bf aw = *(const v16bf*)&swA[(n * 32 + lane) * 16];
#pragma unroll
      for (int fc = 0; fc < 8; ++fc) {
        int c = fc * 16 + ncol;
        const __bf16* src = mh ? zpad : &sF[(n * 128 + c) * 16];
        v16bf bf = *(const v16bf*)src;
        v8f d = {0.f, 0.f, 0.f, 0.f, 0.f, 0.f, 0.f, 0.f};
        d = __builtin_amdgcn_wmma_f32_16x16x32_bf16(
            false, aw, false, bf, (short)0, d, false, false);
        // Scatter wf[n, k, c] into stage-2 A-fragment layout (drop pad row 15).
#pragma unroll
        for (int v = 0; v < 8; ++v) {
          int k = mh * 8 + v;
          if (k < KP) {
            int col  = k * C1 + c;
            int q    = col >> 5, r = col & 31;
            int p2   = r & 1;
            int tp   = (r >> 1) & 3;
            int half = (r >> 3) & 1;
            int a    = (r >> 4) & 1;
            int e    = ((a << 2) + tp) * 2 + p2;
            int lane2 = (half << 4) + n;
            sA[(q * 32 + lane2) * 16 + e] = (__bf16)d[v];
          }
        }
      }
    }
  }
  __syncthreads();

  // ---- 2: 16x1920 @ 1920x128 GEMM via WMMA, wave w -> f-tile w ----
  {
    int wave = t >> 5, lane = t & 31;
    v8f acc = {0.f, 0.f, 0.f, 0.f, 0.f, 0.f, 0.f, 0.f};
    for (int q = 0; q < NQ; ++q) {
      v16bf av = *(const v16bf*)&sA[(q * 32 + lane) * 16];
      v16bf bv = *(const v16bf*)&Bsw[(((size_t)q * 8 + wave) * 32 + lane) * 16];
      acc = __builtin_amdgcn_wmma_f32_16x16x32_bf16(
          false, av, false, bv, (short)0, acc, false, false);
    }
    int ncol = lane & 15, mh = lane >> 4;
#pragma unroll
    for (int v = 0; v < 8; ++v) {
      int m = mh * 8 + v;
      float r = fmaxf(acc[v], 0.0f);                         // ReLU
      out[((size_t)(row0 + m)) * OUTC + wave * 16 + ncol] = r;
    }
  }
}

// --------------------------- kernel 4: skip concat -------------------------
__global__ __launch_bounds__(256) void skip_kernel(const float* __restrict__ feat2,
                                                   float* __restrict__ out) {
  int t = blockIdx.x * 256 + threadIdx.x;
  if (t >= ROWS * C2) return;
  int row = t / C2, c = t % C2;
  out[(size_t)row * OUTC + FO + c] = feat2[t];
}

// ---------------------------------------------------------------------------
extern "C" void kernel_launch(void* const* d_in, const int* in_sizes, int n_in,
                              void* d_out, int out_size, void* d_ws, size_t ws_size,
                              hipStream_t stream) {
  const float* xyz1  = (const float*)d_in[0];
  const float* feat1 = (const float*)d_in[1];
  const float* xyz2  = (const float*)d_in[2];
  const float* feat2 = (const float*)d_in[3];
  const float* kpts  = (const float*)d_in[4];
  const float* W     = (const float*)d_in[5];
  float* out = (float*)d_out;

  char*   ws  = (char*)d_ws;
  __bf16* Bsw = (__bf16*)ws;                         // 491,520 B
  int*    idx = (int*)(ws + 512 * 1024);             // 2 MB (32768*16 ints)

  prepw_kernel<<<(NQ * 8 * 32 * 16 + 255) / 256, 256, 0, stream>>>(W, Bsw);

  dim3 gknn(N2 / 256, BN);
  knn_kernel<<<gknn, 256, 0, stream>>>(xyz1, xyz2, idx);

  kpconv_kernel<<<ROWS / 16, 256, 0, stream>>>(xyz1, feat1, xyz2, kpts, idx,
                                               Bsw, out);

  skip_kernel<<<(ROWS * C2 + 255) / 256, 256, 0, stream>>>(feat2, out);
}
